// BiMambaVision_77206332113402
// MI455X (gfx1250) — matrix operand order repaired and checked by reading, not compiled, not run on hardware
//
#include <hip/hip_runtime.h>

// ---------------------------------------------------------------------------
// CDNA5 (gfx1250) implementation of BiMambaVision forward.
// All dense math is routed through v_wmma_f32_16x16x32_f16 (wave32 WMMA).
// Entire weight set (~150MB fp32 / ~75MB f16) fits in the 192MB L2, so the
// GEMM streams A/B straight from global (L2-resident) without LDS staging.
// Fast path: 2x4 register-blocked tiles (32x64 C per wave), double-buffered
// fragments (copy-free rotation, peeled tail) so global_load_b128 overlaps
// the 8-WMMA bursts without spilling.
// ---------------------------------------------------------------------------

typedef __attribute__((ext_vector_type(16))) _Float16 v16h;
typedef __attribute__((ext_vector_type(8)))  _Float16 v8h;
typedef __attribute__((ext_vector_type(8)))  float    v8f;

union V16U { v16h v; v8h h[2]; };

__device__ __forceinline__ v8f wmma32(const V16U& a, const V16U& b, v8f c) {
    return __builtin_amdgcn_wmma_f32_16x16x32_f16(false, a.v, false, b.v,
                                                  (short)0, c, false, false);
}

// ---------------------------------------------------------------------------
// Fast WMMA GEMM: requires M%32==0, N%64==0, K%64==0 (operands zero-padded).
// One wave computes a 32x64 C block: 2 A-frags x 4 B-frags -> 8 WMMA / k-step.
// ---------------------------------------------------------------------------
__attribute__((amdgpu_waves_per_eu(1)))
__global__ void __launch_bounds__(256)
k_wmma_gemm_fast(const _Float16* __restrict__ A,
                 const _Float16* __restrict__ Wt,
                 const float* __restrict__ bias,
                 float* __restrict__ C,
                 int M, int N, int K, int act)
{
    const int lane = threadIdx.x & 31;
    const int wave = threadIdx.x >> 5;
    const int tiles_n = N >> 6;
    const int tiles_m = M >> 5;
    const int tile = blockIdx.x * 8 + wave;
    if (tile >= tiles_m * tiles_n) return;          // uniform per wave
    const int tm = tile / tiles_n;
    const int tn = tile % tiles_n;
    const int r  = lane & 15;
    const int g  = lane >> 4;
    const size_t sK = (size_t)K;

    const _Float16* a0 = A  + (size_t)(tm * 32 + r) * sK + g * 16;       // rows r, r+16
    const _Float16* b0 = Wt + (size_t)(tn * 64 + r) * sK + g * 16;       // cols r + 16*ni

    V16U af0[2], bf0[4], af1[2], bf1[4];
    v8f  acc[2][4] = {};

#define LOADF(af, bf, k0)                                                     \
    af[0].h[0] = *(const v8h*)(a0 + (k0));                                    \
    af[0].h[1] = *(const v8h*)(a0 + (k0) + 8);                                \
    af[1].h[0] = *(const v8h*)(a0 + 16 * sK + (k0));                          \
    af[1].h[1] = *(const v8h*)(a0 + 16 * sK + (k0) + 8);                      \
    bf[0].v    = *(const v16h*)(b0 + (k0));                                   \
    bf[1].v    = *(const v16h*)(b0 + 16 * sK + (k0));                         \
    bf[2].v    = *(const v16h*)(b0 + 32 * sK + (k0));                         \
    bf[3].v    = *(const v16h*)(b0 + 48 * sK + (k0));

#define WMMA8(af, bf)                                                         \
    _Pragma("unroll") for (int mi = 0; mi < 2; ++mi)                          \
    _Pragma("unroll") for (int ni = 0; ni < 4; ++ni)                          \
        acc[mi][ni] = wmma32(af[mi], bf[ni], acc[mi][ni]);

    LOADF(af0, bf0, 0);
    int k0 = 0;
    for (; k0 + 64 < K; k0 += 64) {                 // copy-free rotation
        LOADF(af1, bf1, k0 + 32);
        WMMA8(af0, bf0);
        LOADF(af0, bf0, k0 + 64);
        WMMA8(af1, bf1);
    }
    LOADF(af1, bf1, k0 + 32);                       // peeled tail (K%64==0)
    WMMA8(af0, bf0);
    WMMA8(af1, bf1);
#undef LOADF
#undef WMMA8

#pragma unroll
    for (int ni = 0; ni < 4; ++ni) {
        const int col = tn * 64 + ni * 16 + r;
        const float bv = bias ? bias[col] : 0.0f;
#pragma unroll
        for (int mi = 0; mi < 2; ++mi) {
#pragma unroll
            for (int i = 0; i < 8; ++i) {
                const int m = tm * 32 + mi * 16 + g * 8 + i;
                float v = acc[mi][ni][i] + bv;
                if (act == 1)      v = fmaxf(v, 0.0f);
                else if (act == 3) v = (v > 20.0f) ? v : log1pf(__expf(v));
                C[(size_t)m * N + col] = v;
            }
        }
    }
}

// ---------------------------------------------------------------------------
// Generic WMMA GEMM (ragged shapes):  C[M,N] = A[M,K] * Wt[N,K]^T
// A rows padded to 16, K padded to 32. act: 0 none, 1 relu, 3 softplus.
// ---------------------------------------------------------------------------
__global__ void k_wmma_gemm(const _Float16* __restrict__ A,
                            const _Float16* __restrict__ Wt,
                            const float* __restrict__ bias,
                            float* __restrict__ C,
                            int M, int N, int K, int act)
{
    const int lane = threadIdx.x & 31;
    const int wave = threadIdx.x >> 5;
    const int tiles_n = (N + 15) >> 4;
    const int tiles_m = (M + 15) >> 4;
    const int tile = blockIdx.x * 8 + wave;
    if (tile >= tiles_m * tiles_n) return;
    const int tm = tile / tiles_n;
    const int tn = tile % tiles_n;
    const int r  = lane & 15;
    const int g  = lane >> 4;
    const size_t sK = (size_t)K;
    const _Float16* arow = A  + (size_t)(tm * 16 + r) * sK + g * 16;
    const _Float16* brow = Wt + (size_t)(tn * 16 + r) * sK + g * 16;

    v8f acc = {};
    for (int k0 = 0; k0 < K; k0 += 32) {
        if (k0 + 64 < K) {
            __builtin_prefetch(arow + k0 + 64, 0, 3);   // near-scope prefetch
            __builtin_prefetch(brow + k0 + 64, 0, 3);
        }
        V16U a, b;
        a.h[0] = *(const v8h*)(arow + k0);
        a.h[1] = *(const v8h*)(arow + k0 + 8);
        b.v    = *(const v16h*)(brow + k0);
        acc = wmma32(a, b, acc);
    }

    const int col = tn * 16 + r;
    if (col >= N) return;
    const float bv = bias ? bias[col] : 0.0f;
#pragma unroll
    for (int i = 0; i < 8; ++i) {
        const int m = tm * 16 + g * 8 + i;
        if (m < M) {
            float v = acc[i] + bv;
            if (act == 1)      v = fmaxf(v, 0.0f);
            else if (act == 3) v = (v > 20.0f) ? v : log1pf(__expf(v));
            C[(size_t)m * N + col] = v;
        }
    }
}

// ---------------------------------------------------------------------------
// Elementwise / data-movement kernels
// ---------------------------------------------------------------------------
__global__ void k_cast_pad(const float* __restrict__ src, long ld, long col0,
                           _Float16* __restrict__ dst, int M, int N, int Np,
                           long total)
{
    long i = (long)blockIdx.x * 256 + threadIdx.x;
    if (i >= total) return;
    int c = (int)(i % Np);
    long r = i / Np;
    float v = (r < M && c < N) ? src[r * ld + col0 + c] : 0.0f;
    dst[i] = (_Float16)v;
}

__global__ void k_nchw_nhwc(const float* __restrict__ src, float* __restrict__ dst,
                            int C, int H, int W, long total)
{
    long i = (long)blockIdx.x * 256 + threadIdx.x;
    if (i >= total) return;
    int c = (int)(i % C); long t = i / C;
    int w = (int)(t % W); t /= W;
    int h = (int)(t % H); long bn = t / H;
    dst[i] = src[(((size_t)bn * C + c) * H + h) * W + w];
}

__global__ void k_im2col2d(const float* __restrict__ x /*NHWC*/, _Float16* __restrict__ dst,
                           int img0, int C, int H, int W, int KH, int KW, int pad,
                           int Kp, long total)
{
    long i = (long)blockIdx.x * 256 + threadIdx.x;
    if (i >= total) return;
    int k = (int)(i % Kp);
    long row = i / Kp;
    int ow = (int)(row % W); long r2 = row / W;
    int oh = (int)(r2 % H); int img = (int)(r2 / H);
    float v = 0.0f;
    if (k < C * KH * KW) {
        int kw = k % KW, kh = (k / KW) % KH, c = k / (KW * KH);
        int ih = oh + kh - pad, iw = ow + kw - pad;
        if (ih >= 0 && ih < H && iw >= 0 && iw < W)
            v = x[(((size_t)(img0 + img) * H + ih) * W + iw) * C + c];
    }
    dst[i] = (_Float16)v;
}

__global__ void k_im2col1d(const float* __restrict__ x /*[B,L,C]*/, _Float16* __restrict__ dst,
                           int L, int C, int KW, int pad, int Kp, long total)
{
    long i = (long)blockIdx.x * 256 + threadIdx.x;
    if (i >= total) return;
    int k = (int)(i % Kp);
    long row = i / Kp;
    int t = (int)(row % L); long b = row / L;
    float v = 0.0f;
    if (k < C * KW) {
        int kw = k % KW, c = k / KW;
        int ts = t + kw - pad;
        if (ts >= 0 && ts < L) v = x[((size_t)b * L + ts) * C + c];
    }
    dst[i] = (_Float16)v;
}

__global__ void k_pool_flat(const float* __restrict__ x /*[BN,8,9,64] NHWC*/,
                            float* __restrict__ dst /*[BN,1024] CHW order*/,
                            long total)
{
    long i = (long)blockIdx.x * 256 + threadIdx.x;
    if (i >= total) return;
    int f = (int)(i % 1024); long bn = i / 1024;
    int pw = f % 4, ph = (f / 4) % 4, c = f / 16;
    float m = -3.4e38f;
#pragma unroll
    for (int dh = 0; dh < 2; ++dh)
#pragma unroll
        for (int dw = 0; dw < 2; ++dw) {
            float v = x[(((size_t)bn * 8 + ph * 2 + dh) * 9 + pw * 2 + dw) * 64 + c];
            m = fmaxf(m, v);
        }
    dst[i] = m;
}

__global__ void k_bn_relu_add(const float* __restrict__ conv, const float* __restrict__ skip,
                              const float* __restrict__ g, const float* __restrict__ b,
                              const float* __restrict__ mean, const float* __restrict__ var,
                              float* __restrict__ out, int C, long total)
{
    long i = (long)blockIdx.x * 256 + threadIdx.x;
    if (i >= total) return;
    int c = (int)(i % C);
    float v = (conv[i] - mean[c]) * rsqrtf(var[c] + 1e-5f) * g[c] + b[c];
    out[i] = fmaxf(v, 0.0f) + skip[i];
}

__global__ void k_concat(const float* __restrict__ a, const float* __restrict__ b,
                         float* __restrict__ o, int d, long total)
{
    long i = (long)blockIdx.x * 256 + threadIdx.x;
    if (i >= total) return;
    int c = (int)(i % (2 * d)); long row = i / (2 * d);
    o[i] = (c < d) ? a[row * d + c] : b[row * d + (c - d)];
}

// depthwise causal conv (D_CONV=4, left pad 3) + bias + SiLU; dual f32/f16 out
__global__ void k_dwconv(const float* __restrict__ xz /*[M,2di]*/,
                         const float* __restrict__ w /*[di,4]*/,
                         const float* __restrict__ bias,
                         float* __restrict__ xc32, _Float16* __restrict__ xc16,
                         int di, int L, long total)
{
    long i = (long)blockIdx.x * 256 + threadIdx.x;
    if (i >= total) return;
    int d = (int)(i % di); long row = i / di;
    int t = (int)(row % L); long b = row / L;
    const float* wr = w + (size_t)d * 4;
    float s = bias[d];
#pragma unroll
    for (int k = 0; k < 4; ++k) {
        int ts = t + k - 3;
        if (ts >= 0)
            s += xz[((size_t)(b * L + ts)) * (2 * (size_t)di) + d] * wr[k];
    }
    s = s / (1.0f + __expf(-s));      // silu
    xc32[i] = s;
    xc16[i] = (_Float16)s;
}

// selective scan: one thread per (batch, channel); h[16] in registers
__global__ void k_scan(const float* __restrict__ dt, const float* __restrict__ xc,
                       const float* __restrict__ dbc, int dtr, int dbw,
                       const float* __restrict__ A_log, const float* __restrict__ Dp,
                       const float* __restrict__ xz, float* __restrict__ y,
                       int di, int L)
{
    int tid = blockIdx.x * 256 + threadIdx.x;
    if (tid >= 4 * di) return;
    int b = tid / di, d = tid % di;
    float a[16], h[16];
#pragma unroll
    for (int n = 0; n < 16; ++n) { a[n] = -__expf(A_log[(size_t)d * 16 + n]); h[n] = 0.0f; }
    const float Dv = Dp[d];
    for (int t = 0; t < L; ++t) {
        size_t row = (size_t)b * L + t;
        float dtv = dt[row * di + d];
        float xv  = xc[row * di + d];
        const float* db = dbc + row * dbw + dtr;   // B at [0..15], C at [16..31]
        float acc = 0.0f;
#pragma unroll
        for (int n = 0; n < 16; ++n) {
            h[n] = __expf(dtv * a[n]) * h[n] + dtv * db[n] * xv;
            acc += h[n] * db[16 + n];
        }
        acc += xv * Dv;
        float zv = xz[row * (size_t)(2 * di) + di + d];
        acc *= zv / (1.0f + __expf(-zv));
        y[row * di + d] = acc;
    }
}

// out = resid + LayerNorm(x)*g + b  (block per row; out may alias resid)
__global__ void k_ln_res(const float* __restrict__ resid, const float* __restrict__ x,
                         const float* __restrict__ g, const float* __restrict__ b,
                         float* __restrict__ out, int d)
{
    __shared__ float s1[256], s2[256];
    const size_t row = blockIdx.x;
    const float* xr = x + row * d;
    float s = 0.0f, q = 0.0f;
    for (int i = threadIdx.x; i < d; i += 256) { float v = xr[i]; s += v; q += v * v; }
    s1[threadIdx.x] = s; s2[threadIdx.x] = q;
    __syncthreads();
    for (int o = 128; o > 0; o >>= 1) {
        if ((int)threadIdx.x < o) { s1[threadIdx.x] += s1[threadIdx.x + o]; s2[threadIdx.x] += s2[threadIdx.x + o]; }
        __syncthreads();
    }
    float mean = s1[0] / d;
    float var  = s2[0] / d - mean * mean;
    float inv  = rsqrtf(var + 1e-6f);
    for (int i = threadIdx.x; i < d; i += 256)
        out[row * d + i] = resid[row * d + i] + (xr[i] - mean) * inv * g[i] + b[i];
}

// gather last token of each batch into padded f16 matrix [16, d]
__global__ void k_last16(const float* __restrict__ x, _Float16* __restrict__ dst,
                         int L, int d, long total)
{
    long i = (long)blockIdx.x * 256 + threadIdx.x;
    if (i >= total) return;
    int rr = (int)(i / d), c = (int)(i % d);
    dst[i] = (rr < 4) ? (_Float16)x[((size_t)rr * L + L - 1) * d + c] : (_Float16)0.0f;
}

// ---------------------------------------------------------------------------
// Host-side orchestration
// ---------------------------------------------------------------------------
namespace {

struct Arena { char* base; size_t off; };
static void* aalloc(Arena& a, size_t bytes) {
    size_t o = (a.off + 255) & ~(size_t)255;
    a.off = o + bytes;
    return a.base + o;
}
static inline int up16(int x) { return (x + 15) & ~15; }
static inline int up32(int x) { return (x + 31) & ~31; }
static inline unsigned gblk(long total) { return (unsigned)((total + 255) / 256); }

static void cast_pad(hipStream_t s, const float* src, long ld, long col0,
                     _Float16* dst, int M, int N, int Mp, int Np) {
    long total = (long)Mp * Np;
    k_cast_pad<<<gblk(total), 256, 0, s>>>(src, ld, col0, dst, M, N, Np, total);
}
static void prep_w(hipStream_t s, const float* w, _Float16* dst, int Nout, int K, int Kp) {
    cast_pad(s, w, K, 0, dst, Nout, K, up16(Nout), Kp);
}
static void gemm(hipStream_t s, const _Float16* A, const _Float16* Wt,
                 const float* bias, float* C, int M, int N, int K, int act) {
    if ((M % 32) == 0 && (N % 64) == 0 && (K % 64) == 0) {
        int tiles = (M / 32) * (N / 64);
        k_wmma_gemm_fast<<<(tiles + 7) / 8, 256, 0, s>>>(A, Wt, bias, C, M, N, K, act);
    } else {
        int tiles = ((M + 15) / 16) * ((N + 15) / 16);
        k_wmma_gemm<<<(tiles + 7) / 8, 256, 0, s>>>(A, Wt, bias, C, M, N, K, act);
    }
}

struct CnnW { const float *w[4], *b[4], *fcw, *fcb; };
struct MambaW { const float *in_proj, *conv_w, *conv_b, *x_proj, *dt_w, *dt_b, *A_log, *D, *out_proj; };

static void run_cnn(hipStream_t s, const float* in_nchw, const CnnW& cw,
                    float* feat, Arena ar, _Float16* wbuf) {
    const int BN = 1024, H = 8, W = 9, HW = 72;
    float*     nhwc   = (float*)aalloc(ar, (size_t)BN * HW * 4 * 4);
    float*     bufA   = (float*)aalloc(ar, (size_t)BN * HW * 256 * 4);
    float*     bufB   = (float*)aalloc(ar, (size_t)BN * HW * 256 * 4);
    _Float16*  col    = (_Float16*)aalloc(ar, (size_t)9216 * 1152 * 2);
    float*     flat   = (float*)aalloc(ar, (size_t)BN * 1024 * 4);
    _Float16*  flat16 = (_Float16*)aalloc(ar, (size_t)BN * 1024 * 2);

    { long t = (long)BN * 4 * HW; k_nchw_nhwc<<<gblk(t), 256, 0, s>>>(in_nchw, nhwc, 4, H, W, t); }

    auto conv = [&](const float* xin, const float* wt, const float* bs, float* xout,
                    int Cin, int Cout, int KH, int KW, int pad) {
        int K = Cin * KH * KW, Kp = up32(K);
        prep_w(s, wt, wbuf, Cout, K, Kp);
        for (int c = 0; c < 8; ++c) {              // 128-image chunks
            int img0 = c * 128;
            long t = (long)9216 * Kp;
            k_im2col2d<<<gblk(t), 256, 0, s>>>(xin, col, img0, Cin, H, W, KH, KW, pad, Kp, t);
            gemm(s, col, wbuf, bs, xout + (size_t)img0 * HW * Cout, 9216, Cout, Kp, 1);
        }
    };
    conv(nhwc, cw.w[0], cw.b[0], bufA,   4,  64, 5, 5, 2);
    conv(bufA, cw.w[1], cw.b[1], bufB,  64, 128, 3, 3, 1);
    conv(bufB, cw.w[2], cw.b[2], bufA, 128, 256, 3, 3, 1);
    conv(bufA, cw.w[3], cw.b[3], bufB, 256,  64, 1, 1, 0);

    { long t = (long)BN * 1024; k_pool_flat<<<gblk(t), 256, 0, s>>>(bufB, flat, t); }
    cast_pad(s, flat, 1024, 0, flat16, BN, 1024, BN, 1024);
    prep_w(s, cw.fcw, wbuf, 512, 1024, 1024);
    gemm(s, flat16, wbuf, cw.fcb, feat, BN, 512, 1024, 1);
}

static void run_resblock(hipStream_t s, const float* feat, int Cin, int Cout,
                         const float* convw, const float* bng, const float* bnb,
                         const float* bnm, const float* bnv,
                         const float* skipw, const float* skip_feat,
                         float* out, Arena ar, _Float16* wbuf) {
    const int M = 1024, L = 256;
    int K = Cin * 3;                                // 1536 / 6144, mult of 64
    _Float16* col  = (_Float16*)aalloc(ar, (size_t)M * K * 2);
    float*    conv = (float*)aalloc(ar, (size_t)M * Cout * 4);
    { long t = (long)M * K; k_im2col1d<<<gblk(t), 256, 0, s>>>(feat, col, L, Cin, 3, 1, K, t); }
    prep_w(s, convw, wbuf, Cout, K, K);
    gemm(s, col, wbuf, nullptr, conv, M, Cout, K, 0);
    const float* skip = skip_feat;
    if (skipw) {
        _Float16* f16 = (_Float16*)aalloc(ar, (size_t)M * Cin * 2);
        float*    sk  = (float*)aalloc(ar, (size_t)M * Cout * 4);
        cast_pad(s, feat, Cin, 0, f16, M, Cin, M, Cin);
        prep_w(s, skipw, wbuf, Cout, Cin, Cin);
        gemm(s, f16, wbuf, nullptr, sk, M, Cout, Cin, 0);
        skip = sk;
    }
    long t2 = (long)M * Cout;
    k_bn_relu_add<<<gblk(t2), 256, 0, s>>>(conv, skip, bng, bnb, bnm, bnv, out, Cout, t2);
}

static void run_mamba(hipStream_t s, const float* xin, float* mout, int d,
                      const MambaW& w, Arena ar, _Float16* wbuf) {
    const int M = 1024, L = 256, di = 2 * d, dtr = d / 16, dbw = dtr + 32;
    _Float16* a16   = (_Float16*)aalloc(ar, (size_t)M * d * 2);
    float*    xz    = (float*)aalloc(ar, (size_t)M * 2 * di * 4);
    float*    xc32  = (float*)aalloc(ar, (size_t)M * di * 4);
    _Float16* xc16  = (_Float16*)aalloc(ar, (size_t)M * di * 2);
    float*    dbc   = (float*)aalloc(ar, (size_t)M * dbw * 4);
    _Float16* dbc16 = (_Float16*)aalloc(ar, (size_t)M * dtr * 2);
    float*    dt    = (float*)aalloc(ar, (size_t)M * di * 4);
    float*    y     = (float*)aalloc(ar, (size_t)M * di * 4);
    _Float16* y16   = (_Float16*)aalloc(ar, (size_t)M * di * 2);

    cast_pad(s, xin, d, 0, a16, M, d, M, d);
    prep_w(s, w.in_proj, wbuf, 2 * di, d, d);
    gemm(s, a16, wbuf, nullptr, xz, M, 2 * di, d, 0);

    { long t = (long)M * di; k_dwconv<<<gblk(t), 256, 0, s>>>(xz, w.conv_w, w.conv_b, xc32, xc16, di, L, t); }

    prep_w(s, w.x_proj, wbuf, dbw, di, di);
    gemm(s, xc16, wbuf, nullptr, dbc, M, dbw, di, 0);

    cast_pad(s, dbc, dbw, 0, dbc16, M, dtr, M, dtr);
    prep_w(s, w.dt_w, wbuf, di, dtr, dtr);
    gemm(s, dbc16, wbuf, w.dt_b, dt, M, di, dtr, 3);      // fused softplus

    { int tot = 4 * di; k_scan<<<gblk(tot), 256, 0, s>>>(dt, xc32, dbc, dtr, dbw,
                                                         w.A_log, w.D, xz, y, di, L); }

    cast_pad(s, y, di, 0, y16, M, di, M, di);
    prep_w(s, w.out_proj, wbuf, d, di, di);
    gemm(s, y16, wbuf, nullptr, mout, M, d, di, 0);
}

} // namespace

// ---------------------------------------------------------------------------
// Input index map (setup_inputs depth-first dict order):
//  0 g, 1 r
//  2..11  cnn_g: c1w c1b c2w c2b c3w c3b c4w c4b fcw fcb
// 12..21  cnn_r
// 22..33  co_res: e_conv, e_bn{g,b,mean,var}, g_conv, g_bn{g,b,mean,var}, e_skip, g_skip
// 34..42  co_mamba_g: in_proj conv_w conv_b x_proj dt_w dt_b A_log D out_proj
// 43..51  co_mamba_r
// 52..55  co_ln1_g co_ln1_b co_ln2_g co_ln2_b
// 56..60  en_conv, en_bn{g,b,mean,var}
// 61..69  en_mamba
// 70..71  en_ln_g en_ln_b
// 72..73  cls_w cls_b
// ---------------------------------------------------------------------------
extern "C" void kernel_launch(void* const* d_in, const int* in_sizes, int n_in,
                              void* d_out, int out_size, void* d_ws, size_t ws_size,
                              hipStream_t stream)
{
    (void)in_sizes; (void)n_in; (void)out_size; (void)ws_size;
    auto P = [&](int i) { return (const float*)d_in[i]; };
    const float* g = P(0);
    const float* r = P(1);

    CnnW cg { { P(2), P(4), P(6), P(8) },  { P(3), P(5), P(7), P(9) },  P(10), P(11) };
    CnnW cr { { P(12), P(14), P(16), P(18) }, { P(13), P(15), P(17), P(19) }, P(20), P(21) };
    MambaW mg { P(34), P(35), P(36), P(37), P(38), P(39), P(40), P(41), P(42) };
    MambaW mr { P(43), P(44), P(45), P(46), P(47), P(48), P(49), P(50), P(51) };
    MambaW me { P(61), P(62), P(63), P(64), P(65), P(66), P(67), P(68), P(69) };

    Arena base { (char*)d_ws, 0 };
    float* gf   = (float*)aalloc(base, (size_t)1024 * 512 * 4);
    float* rf   = (float*)aalloc(base, (size_t)1024 * 512 * 4);
    float* g1   = (float*)aalloc(base, (size_t)1024 * 1024 * 4);
    float* r1   = (float*)aalloc(base, (size_t)1024 * 1024 * 4);
    float* xcat = (float*)aalloc(base, (size_t)1024 * 2048 * 4);
    float* x2   = (float*)aalloc(base, (size_t)1024 * 2048 * 4);
    float* mtmp = (float*)aalloc(base, (size_t)1024 * 2048 * 4);
    _Float16* wbuf  = (_Float16*)aalloc(base, (size_t)8192 * 2048 * 2);   // largest f16 weight
    _Float16* cls16 = (_Float16*)aalloc(base, (size_t)16 * 2048 * 2);

    // --- CNN towers (scratch arenas passed by value -> auto reset) ---
    run_cnn(stream, g, cg, gf, base, wbuf);
    run_cnn(stream, r, cr, rf, base, wbuf);

    // --- conv1d residual blocks ---
    run_resblock(stream, gf, 512, 1024, P(22), P(23), P(24), P(25), P(26), P(32), nullptr, g1, base, wbuf);
    run_resblock(stream, rf, 512, 1024, P(27), P(28), P(29), P(30), P(31), P(33), nullptr, r1, base, wbuf);

    // --- co mamba blocks + LN residual (in place) ---
    run_mamba(stream, g1, mtmp, 1024, mg, base, wbuf);
    k_ln_res<<<1024, 256, 0, stream>>>(g1, mtmp, P(52), P(53), g1, 1024);
    run_mamba(stream, r1, mtmp, 1024, mr, base, wbuf);
    k_ln_res<<<1024, 256, 0, stream>>>(r1, mtmp, P(54), P(55), r1, 1024);

    // --- concat + en residual conv block (identity skip) ---
    { long t = (long)1024 * 2048; k_concat<<<gblk(t), 256, 0, stream>>>(g1, r1, xcat, 1024, t); }
    run_resblock(stream, xcat, 2048, 2048, P(56), P(57), P(58), P(59), P(60), nullptr, xcat, x2, base, wbuf);

    // --- en mamba + LN residual ---
    run_mamba(stream, x2, mtmp, 2048, me, base, wbuf);
    k_ln_res<<<1024, 256, 0, stream>>>(x2, mtmp, P(70), P(71), x2, 2048);

    // --- classifier on last token ---
    { long t = (long)16 * 2048; k_last16<<<gblk(t), 256, 0, stream>>>(x2, cls16, 256, 2048, t); }
    prep_w(stream, P(72), wbuf, 7, 2048, 2048);
    gemm(stream, cls16, wbuf, P(73), (float*)d_out, 4, 7, 2048, 0);
}
